// Attention_87393994539544
// MI455X (gfx1250) — matrix operand order repaired and checked by reading, not compiled, not run on hardware
//
#include <hip/hip_runtime.h>

// ---------------------------------------------------------------------------
// Attention for B=8, S=4096, D=96 (fp32 in/out) on gfx1250 (MI455X).
// Compute-bound (~53 GFLOP vs ~50MB HBM traffic) -> bf16 WMMA, f32 accumulate.
// Kernel 0a/0b: one-time f32->bf16 conversion of h and W (log2e/sqrt(D)
//               folded into Wq so softmax runs in exp2 domain).
// Kernel 1: QKV projection (WMMA) from pre-converted bf16 operands.
//           V stored transposed so PV B-operands load contiguously.
// Kernel 2: flash attention, 1 wave per 16-query tile, 32-key steps,
//           12 v_wmma_f32_16x16x32_bf16 per step + online softmax.
// ---------------------------------------------------------------------------

typedef __attribute__((ext_vector_type(16))) __bf16 v16bf;
typedef __attribute__((ext_vector_type(2)))  __bf16 v2bf;
typedef __attribute__((ext_vector_type(8)))  float  v8f;

#define DHEAD 96
#define SLEN  4096
#define BATCH 8
#define QSCALE (1.4426950408889634f * 0.10206207261596577f)  // log2e/sqrt(96)

union ABOp { unsigned u[8]; v16bf v; };

// Packed f32x2 -> bf16x2: prefer hw cvt; else round-half-up + one v_perm_b32.
__device__ __forceinline__ unsigned pack2(float lo, float hi) {
#if __has_builtin(__builtin_amdgcn_cvt_pk_bf16_f32)
  union { v2bf v; unsigned u; } x;
  x.v = __builtin_amdgcn_cvt_pk_bf16_f32(lo, hi);
  return x.u;
#else
  union { float f; unsigned u; } a, b; a.f = lo; b.f = hi;
  return __builtin_amdgcn_perm(b.u + 0x8000u, a.u + 0x8000u, 0x07060302u);
#endif
}
__device__ __forceinline__ unsigned short f2bf(float f) {
  union { float f; unsigned u; } x; x.f = f;
  return (unsigned short)((x.u + 0x8000u) >> 16);
}
// 16-bit A-operand (16xK=32): VGPR v packs K = 16*half + {2v,2v+1}
// (contiguous order, matching the B-operand fill).
__device__ __forceinline__ int a_kbase(int v, int half) {
  return 16 * half + 2 * v;
}
// 16-bit B-operand (K=32 x 16): lane n = lane%16; VGPR v packs K = 16*half+2v
// -> a lane's 8 VGPRs are 16 contiguous K elements (32 contiguous bytes).

// ---------------------------------------------------------------------------
// Kernel 0a: h (f32) -> hbf (bf16), 8 elements per thread, b128 in / b128 out.
// ---------------------------------------------------------------------------
__global__ void __launch_bounds__(256)
convert_h_kernel(const float4* __restrict__ h, uint4* __restrict__ hb) {
  const size_t i = (size_t)blockIdx.x * blockDim.x + threadIdx.x;
  float4 x0 = h[2 * i], x1 = h[2 * i + 1];
  uint4 r;
  r.x = pack2(x0.x, x0.y);
  r.y = pack2(x0.z, x0.w);
  r.z = pack2(x1.x, x1.y);
  r.w = pack2(x1.z, x1.w);
  hb[i] = r;
}

// ---------------------------------------------------------------------------
// Kernel 0b: Wq/Wk/Wv (f32) -> bf16, Wq scaled by log2e/sqrt(D). 1 pair/thread.
// ---------------------------------------------------------------------------
__global__ void __launch_bounds__(256)
convert_w_kernel(const float* __restrict__ Wq, const float* __restrict__ Wk,
                 const float* __restrict__ Wv, unsigned* __restrict__ wb) {
  const int i = blockIdx.x * blockDim.x + threadIdx.x;   // pair index, 3*4608
  const int mat = i / (DHEAD * DHEAD / 2);
  const int p   = i % (DHEAD * DHEAD / 2);
  const float* W = (mat == 0) ? Wq : (mat == 1) ? Wk : Wv;
  const float sc = (mat == 0) ? QSCALE : 1.0f;
  wb[i] = pack2(W[2 * p] * sc, W[2 * p + 1] * sc);
}

// ---------------------------------------------------------------------------
// Kernel 1: Q/K/V = h @ W^T + b.   One wave per (16-row, 16-col, matrix) tile.
// All matrix operands pre-converted bf16; B loads are contiguous 32B/lane.
// ---------------------------------------------------------------------------
__global__ void __launch_bounds__(256)
qkv_proj_kernel(const unsigned short* __restrict__ hbf,  // [B*S, 96] bf16
                const unsigned short* __restrict__ Wbf,  // [3, 96, 96] bf16
                const float* __restrict__ bq, const float* __restrict__ bk,
                const float* __restrict__ bv,
                unsigned short* __restrict__ Qb,   // [B*S, 96] bf16 (pre-scaled)
                unsigned short* __restrict__ Kb,   // [B*S, 96] bf16
                unsigned short* __restrict__ Vt)   // [B, 96, S] bf16 (transposed)
{
  const int lane = threadIdx.x & 31;
  const int wave = threadIdx.x >> 5;
  const int task = blockIdx.x * 8 + wave;          // rowTile*18 + mat*6 + ntile
  const int ntile   = task % 6;
  const int mat     = (task / 6) % 3;
  const int rowTile = task / 18;
  const int row0 = rowTile * 16;
  const int m = lane & 15, half = lane >> 4, n = lane & 15;
  const int e0 = ntile * 16;

  const float* bias = (mat == 0) ? bq : (mat == 1) ? bk : bv;
  const unsigned short* W = Wbf + mat * DHEAD * DHEAD;

  // A operands: 16 rows of hbf, K = 96 split into 3 chunks of 32.
  ABOp A[3];
  const unsigned short* hrow = hbf + (size_t)(row0 + m) * DHEAD;
#pragma unroll
  for (int c = 0; c < 3; ++c)
#pragma unroll
    for (int v = 0; v < 8; ++v)
      A[c].u[v] = *(const unsigned*)(hrow + c * 32 + a_kbase(v, half));

  v8f acc = {0.f, 0.f, 0.f, 0.f, 0.f, 0.f, 0.f, 0.f};
#pragma unroll
  for (int c = 0; c < 3; ++c) {
    ABOp Bm;                                       // B[k][n] = W[e0+n][c*32+k]
    const unsigned short* wrow =
        W + (size_t)(e0 + n) * DHEAD + c * 32 + 16 * half;
#pragma unroll
    for (int v = 0; v < 8; ++v)
      Bm.u[v] = *(const unsigned*)(wrow + 2 * v); // merges to 2x b128
    acc = __builtin_amdgcn_wmma_f32_16x16x32_bf16(false, A[c].v, false, Bm.v,
                                                  (short)0, acc, false, false);
  }

  // Bias (scale Wq fold means bq must be scaled identically).
  const float bval = bias[e0 + n] * ((mat == 0) ? QSCALE : 1.0f);

  if (mat == 2) {                                  // V stored transposed
#pragma unroll
    for (int r = 0; r < 8; ++r) {
      int row = row0 + r + 8 * half;
      int b = row >> 12, s = row & (SLEN - 1);
      Vt[((size_t)b * DHEAD + e0 + n) * SLEN + s] = f2bf(acc[r] + bval);
    }
  } else {
    unsigned short* Out = (mat == 0) ? Qb : Kb;
#pragma unroll
    for (int r = 0; r < 8; ++r) {
      int row = row0 + r + 8 * half;
      Out[(size_t)row * DHEAD + e0 + n] = f2bf(acc[r] + bval);
    }
  }
}

// ---------------------------------------------------------------------------
// Kernel 2: flash attention. One wave per 16-query tile; 128 steps of 32 keys.
// ---------------------------------------------------------------------------
#define PSTRIDE 34   // padded row stride (halfwords): 17*m mod 64 banks, no conflict

__global__ void __launch_bounds__(256)
flash_attn_kernel(const unsigned short* __restrict__ Qb,
                  const unsigned short* __restrict__ Kb,
                  const unsigned short* __restrict__ Vt,
                  float* __restrict__ out)
{
  __shared__ unsigned short Psh[8][16 * PSTRIDE];  // per-wave P transpose buffer
  const int lane = threadIdx.x & 31;
  const int wave = threadIdx.x >> 5;
  const int tile = blockIdx.x * 8 + wave;          // 0 .. B*S/16-1
  const int b  = tile >> 8;                        // 256 tiles per batch
  const int q0 = (tile & 255) * 16;
  const int m = lane & 15, half = lane >> 4, n = lane & 15;

  // Q A-operands for the whole tile (3 K-chunks), loaded once.
  ABOp Aq[3];
  const unsigned short* qrow = Qb + ((size_t)b * SLEN + q0 + m) * DHEAD;
#pragma unroll
  for (int c = 0; c < 3; ++c)
#pragma unroll
    for (int v = 0; v < 8; ++v)
      Aq[c].u[v] = *(const unsigned*)(qrow + c * 32 + a_kbase(v, half));

  v8f O[6];
#pragma unroll
  for (int t = 0; t < 6; ++t) O[t] = v8f{0.f, 0.f, 0.f, 0.f, 0.f, 0.f, 0.f, 0.f};
  float mrow[8], lrow[8];
#pragma unroll
  for (int r = 0; r < 8; ++r) { mrow[r] = -__builtin_inff(); lrow[r] = 0.f; }

  const unsigned short* Kbase = Kb + (size_t)b * SLEN * DHEAD;
  const unsigned short* Vbase = Vt + (size_t)b * DHEAD * SLEN;
  unsigned short* Pw = Psh[wave];

  for (int j0 = 0; j0 < SLEN; j0 += 32) {
    if (j0 + 32 < SLEN) {                          // prefetch next K/V tiles
      __builtin_prefetch(Kbase + (size_t)(j0 + 32 + n) * DHEAD, 0, 0);
      __builtin_prefetch(Vbase + (size_t)n * SLEN + j0 + 32, 0, 0);
    }

    // ---- S = Q . K^T (log2e scale folded into Q); two 16-key subtiles.
    //      K-chunk outer so consecutive WMMAs share the A operand. ----
    ABOp Bk[2];
    v8f Sacc[2];
    Sacc[0] = v8f{0.f, 0.f, 0.f, 0.f, 0.f, 0.f, 0.f, 0.f};
    Sacc[1] = v8f{0.f, 0.f, 0.f, 0.f, 0.f, 0.f, 0.f, 0.f};
#pragma unroll
    for (int c = 0; c < 3; ++c) {
#pragma unroll
      for (int t = 0; t < 2; ++t) {
        const unsigned short* p =                  // 32 contiguous bytes/lane
            Kbase + (size_t)(j0 + 16 * t + n) * DHEAD + 16 * half + c * 32;
#pragma unroll
        for (int v = 0; v < 8; ++v)
          Bk[t].u[v] = *(const unsigned*)(p + 2 * v);
      }
#pragma unroll
      for (int t = 0; t < 2; ++t)
        Sacc[t] = __builtin_amdgcn_wmma_f32_16x16x32_bf16(
            false, Aq[c].v, false, Bk[t].v, (short)0, Sacc[t], false, false);
    }

    // ---- online softmax (exp2 domain): row (r,half) spans one 16-lane group
#pragma unroll
    for (int r = 0; r < 8; ++r) {
      float s0 = Sacc[0][r], s1 = Sacc[1][r];
      float vmax = fmaxf(s0, s1);
#pragma unroll
      for (int off = 1; off < 16; off <<= 1)
        vmax = fmaxf(vmax, __shfl_xor(vmax, off, 32));
      float mnew = fmaxf(mrow[r], vmax);
      float corr = __builtin_amdgcn_exp2f(mrow[r] - mnew);
      float p0 = __builtin_amdgcn_exp2f(s0 - mnew);
      float p1 = __builtin_amdgcn_exp2f(s1 - mnew);
      float psum = p0 + p1;
#pragma unroll
      for (int off = 1; off < 16; off <<= 1)
        psum += __shfl_xor(psum, off, 32);
      lrow[r] = lrow[r] * corr + psum;
      mrow[r] = mnew;
#pragma unroll
      for (int t = 0; t < 6; ++t) O[t][r] *= corr;
      int mm = r + 8 * half;                       // C-layout row owned here
      unsigned pp = pack2(p0, p1);                 // one perm + 2 adds
      Pw[mm * PSTRIDE + n]      = (unsigned short)(pp & 0xFFFFu);
      Pw[mm * PSTRIDE + n + 16] = (unsigned short)(pp >> 16);
    }

    // ---- reload P in A-operand layout (transpose through per-wave LDS;
    //      LDS ops are in-order within a wave) ----
    ABOp Ap;
#pragma unroll
    for (int v = 0; v < 8; ++v)
      Ap.u[v] = *(const unsigned*)(Pw + m * PSTRIDE + a_kbase(v, half));

    // ---- O += P . V  (V^T layout -> contiguous K loads per lane) ----
#pragma unroll
    for (int t = 0; t < 6; ++t) {
      ABOp Bv;
      const unsigned short* vrow =
          Vbase + (size_t)(t * 16 + n) * SLEN + j0 + 16 * half;
#pragma unroll
      for (int v = 0; v < 8; ++v)
        Bv.u[v] = *(const unsigned*)(vrow + 2 * v);
      O[t] = __builtin_amdgcn_wmma_f32_16x16x32_bf16(
          false, Ap.v, false, Bv.v, (short)0, O[t], false, false);
    }
  }

  // ---- finalize: out = O / l ----
#pragma unroll
  for (int r = 0; r < 8; ++r) {
    float inv = 1.0f / lrow[r];
    int row = q0 + r + 8 * half;
    float* orow = out + ((size_t)b * SLEN + row) * DHEAD;
#pragma unroll
    for (int t = 0; t < 6; ++t)
      orow[t * 16 + n] = O[t][r] * inv;
  }
}

// ---------------------------------------------------------------------------
extern "C" void kernel_launch(void* const* d_in, const int* in_sizes, int n_in,
                              void* d_out, int out_size, void* d_ws, size_t ws_size,
                              hipStream_t stream) {
  const float* h  = (const float*)d_in[0];
  const float* Wq = (const float*)d_in[1];
  const float* bq = (const float*)d_in[2];
  const float* Wk = (const float*)d_in[3];
  const float* bk = (const float*)d_in[4];
  const float* Wv = (const float*)d_in[5];
  const float* bv = (const float*)d_in[6];
  float* out = (float*)d_out;

  const size_t per = (size_t)BATCH * SLEN * DHEAD;   // elements per matrix
  unsigned short* Qb  = (unsigned short*)d_ws;       // per * 2B each
  unsigned short* Kb  = Qb + per;
  unsigned short* Vt  = Kb + per;
  unsigned short* hbf = Vt + per;                    // bf16 copy of h
  unsigned short* Wbf = hbf + per;                   // [3,96,96] bf16
  // total: (4*per + 3*96*96) * 2B ~ 25.2 MB of workspace

  // 0a: convert h (per elements, 8/thread).
  convert_h_kernel<<<(int)(per / (256 * 8)), 256, 0, stream>>>(
      (const float4*)h, (uint4*)hbf);
  // 0b: convert W (3*96*96/2 pairs, 1 pair/thread).
  convert_w_kernel<<<3 * DHEAD * DHEAD / 2 / 256, 256, 0, stream>>>(
      Wq, Wk, Wv, (unsigned*)Wbf);

  // QKV: (B*S/16 row tiles) * 3 matrices * 6 col tiles = 36864 waves, 8/block.
  const int tilesRows = BATCH * SLEN / 16;           // 2048
  qkv_proj_kernel<<<tilesRows * 18 / 8, 256, 0, stream>>>(
      hbf, Wbf, bq, bk, bv, Qb, Kb, Vt);

  // Flash attention: 2048 query tiles, 8 waves/block -> 256 blocks.
  flash_attn_kernel<<<tilesRows / 8, 256, 0, stream>>>(Qb, Kb, Vt, out);
}
